// Block_61220463837740
// MI455X (gfx1250) — matrix operand order repaired and checked by reading, not compiled, not run on hardware
//
#include <hip/hip_runtime.h>

// ---------------------------------------------------------------------------
// Transformer block for MI455X (gfx1250, wave32, WMMA + async-to-LDS DMA).
// GEMMs: v_wmma_f32_16x16x32_bf16, LDS double-buffered via
// global_load_async_to_lds_b128 (ASYNCcnt). Weights pre-transposed [N][K].
// Attention: flash loop, mask only on the diagonal block, 1/sqrt(HS) folded
// into the Q projection.
// ---------------------------------------------------------------------------

#define B_   2
#define S_   2048
#define D_   1024
#define H_   16
#define HS_  64
#define DFF_ 4096
#define BS_  (B_ * S_)

typedef __attribute__((ext_vector_type(16))) __bf16 v16bf;
typedef __attribute__((ext_vector_type(8)))  float  v8f;

union Frag {
  v16bf v;
  unsigned int u[8];
  unsigned short h[16];
};

__device__ __forceinline__ unsigned short f2bf(float f) {
  unsigned int u = __float_as_uint(f);
  u += 0x7FFFu + ((u >> 16) & 1u);   // round-to-nearest-even
  return (unsigned short)(u >> 16);
}

// A-matrix (16x32 bf16) per-lane K offsets: lanes 0-15 hold K 0..7,16..23;
// lanes 16-31 hold K 8..15,24..31 (pairs packed per VGPR).
__device__ __forceinline__ int kbaseA(int j, int half) {
  return (j < 4 ? 2 * j : 2 * j + 8) + half * 8;
}
// B-matrix (32x16 bf16): same K order as A: lanes 0-15 hold K 0..7,16..23;
// lanes 16-31 hold K 8..15,24..31.
__device__ __forceinline__ int kbaseB(int j, int half) {
  return (j < 4 ? 2 * j : 2 * j + 8) + half * 8;
}

__device__ __forceinline__ v8f wmma_bf16(const Frag& a, const Frag& b, v8f c) {
  return __builtin_amdgcn_wmma_f32_16x16x32_bf16(
      false, a.v, false, b.v, (short)0, c, false, false);
}

// Async DMA: copy 16B per lane from global (va) into LDS (lds byte offset).
__device__ __forceinline__ void async_b128(unsigned lds, unsigned long long va) {
  asm volatile("global_load_async_to_lds_b128 %0, %1, off"
               :: "v"(lds), "v"(va) : "memory");
}
__device__ __forceinline__ void async_b128_o16(unsigned lds, unsigned long long va) {
  // inst offset applies to BOTH lds and global address (ISA 08 §4.4)
  asm volatile("global_load_async_to_lds_b128 %0, %1, off offset:16"
               :: "v"(lds), "v"(va) : "memory");
}

// ---------------------------------------------------------------------------
// Weight converters (run once per launch; negligible vs GEMM time).
// ---------------------------------------------------------------------------
// Wq/Wk/Wv: (H, D, HS) fp32 -> transposed bf16 [N=H*HS][K=D]
__global__ void k_cvt_head(const float* __restrict__ in,
                           unsigned short* __restrict__ out) {
  int i = blockIdx.x * 256 + threadIdx.x;   // over H*D*HS = 1M
  int h   = i >> 16;                        // / (D*HS) = 65536
  int rem = i & 65535;
  int d   = rem >> 6;
  int e   = rem & 63;
  out[((size_t)(h * HS_ + e) << 10) + d] = f2bf(in[i]);   // K = D = 1024
}

// Generic: [K,N] fp32 -> [N,K] bf16 (N = 1<<logN)
__global__ void k_cvt_t(const float* __restrict__ in,
                        unsigned short* __restrict__ out, int Kd, int logN) {
  int i = blockIdx.x * 256 + threadIdx.x;   // over K*N
  int k = i >> logN;
  int n = i & ((1 << logN) - 1);
  out[(size_t)n * Kd + k] = f2bf(in[i]);
}

// ---------------------------------------------------------------------------
// LayerNorm: fp32 row (D=1024) -> bf16 row
// ---------------------------------------------------------------------------
__global__ __launch_bounds__(256)
void k_ln(const float* __restrict__ x, const float* __restrict__ gamma,
          const float* __restrict__ beta, unsigned short* __restrict__ out) {
  __shared__ float red[256];
  const int tid = threadIdx.x;
  const float* xr = x + (size_t)blockIdx.x * D_;
  float v[4];
  float s = 0.f;
#pragma unroll
  for (int i = 0; i < 4; ++i) { v[i] = xr[tid + i * 256]; s += v[i]; }
  red[tid] = s; __syncthreads();
  for (int st = 128; st > 0; st >>= 1) {
    if (tid < st) red[tid] += red[tid + st];
    __syncthreads();
  }
  const float mu = red[0] * (1.f / D_);
  __syncthreads();
  float s2 = 0.f;
#pragma unroll
  for (int i = 0; i < 4; ++i) { float d = v[i] - mu; s2 += d * d; }
  red[tid] = s2; __syncthreads();
  for (int st = 128; st > 0; st >>= 1) {
    if (tid < st) red[tid] += red[tid + st];
    __syncthreads();
  }
  const float rs = rsqrtf(red[0] * (1.f / D_) + 1e-5f);
  unsigned short* orow = out + (size_t)blockIdx.x * D_;
#pragma unroll
  for (int i = 0; i < 4; ++i) {
    int c = tid + i * 256;
    orow[c] = f2bf((v[i] - mu) * rs * gamma[c] + beta[c]);
  }
}

// ---------------------------------------------------------------------------
// WMMA GEMM, async double-buffered LDS staging.
//   C[M,N] = epilogue(A[M,K] * B[K,N] + bias [+ res])
// A bf16 row-major [M][K]; B supplied TRANSPOSED bf16 [N][K].
// Block: 256 thr = 8 waves; tile 128x64; K-step 32; waves 4(rows) x 2(cols).
// EPI: 0 (bias)*scale->bf16 | 1 bias+res->f32 | 2 bias+GELU->bf16
//      3 bias->bf16 scattered to Vt[B,H,HS,S] (head-transposed V)
// ---------------------------------------------------------------------------
template <int EPI>
__global__ __launch_bounds__(256)
void k_gemm(const unsigned short* __restrict__ A,
            const unsigned short* __restrict__ BT,
            const float* __restrict__ bias,
            const float* __restrict__ res,
            void* __restrict__ Cout, int M, int N, int K, float scale) {
  __shared__ __align__(16) unsigned short As[2][128][32];  // 16 KB
  __shared__ __align__(16) unsigned short Bs[2][64][32];   //  8 KB

  const int tid  = threadIdx.x;
  const int wave = tid >> 5;
  const int lane = tid & 31;
  const int half = lane >> 4;
  const int r    = lane & 15;
  const int wm   = wave & 3;
  const int wn   = wave >> 2;
  const int blockRow = blockIdx.y * 128;
  const int blockCol = blockIdx.x * 64;

  v8f acc[2][2];
#pragma unroll
  for (int a = 0; a < 2; ++a)
#pragma unroll
    for (int b = 0; b < 2; ++b)
      acc[a][b] = (v8f){0.f, 0.f, 0.f, 0.f, 0.f, 0.f, 0.f, 0.f};

  // Per-thread DMA slices: A tile 128x32 (32B/thread), B tile 64x32 (16B/thread)
  const int arow = tid >> 1, acol = (tid & 1) * 16;
  const int bn   = tid >> 2, bk   = (tid & 3) * 8;
  const unsigned short* gA = A  + (size_t)(blockRow + arow) * K + acol;
  const unsigned short* gB = BT + (size_t)(blockCol + bn)   * K + bk;

  unsigned lA[2], lB[2];
  lA[0] = (unsigned)(size_t)&As[0][arow][acol];
  lA[1] = (unsigned)(size_t)&As[1][arow][acol];
  lB[0] = (unsigned)(size_t)&Bs[0][bn][bk];
  lB[1] = (unsigned)(size_t)&Bs[1][bn][bk];

  // 3 async instructions per wave per tile (ASYNCcnt += 3)
  auto issue = [&](int k0, int buf) {
    unsigned long long va = (unsigned long long)(size_t)(gA + k0);
    unsigned long long vb = (unsigned long long)(size_t)(gB + k0);
    async_b128(lA[buf], va);
    async_b128_o16(lA[buf], va);
    async_b128(lB[buf], vb);
  };

  issue(0, 0);
  int buf = 0;
  for (int k0 = 0; k0 < K; k0 += 32, buf ^= 1) {
    if (k0 + 32 < K) {
      issue(k0 + 32, buf ^ 1);
      // in-order completion: <=3 in flight means current tile has landed
      asm volatile("s_wait_asynccnt 0x3" ::: "memory");
    } else {
      asm volatile("s_wait_asynccnt 0x0" ::: "memory");
    }
    __syncthreads();  // all waves' DMA for this buffer done

    Frag af[2], bfr[2];
#pragma unroll
    for (int tm = 0; tm < 2; ++tm)
#pragma unroll
      for (int j = 0; j < 8; ++j)
        af[tm].u[j] =
            *(const unsigned int*)&As[buf][wm * 32 + tm * 16 + r][kbaseA(j, half)];
#pragma unroll
    for (int tn = 0; tn < 2; ++tn)
#pragma unroll
      for (int j = 0; j < 8; ++j)
        bfr[tn].u[j] =
            *(const unsigned int*)&Bs[buf][wn * 32 + tn * 16 + r][kbaseB(j, half)];
#pragma unroll
    for (int tm = 0; tm < 2; ++tm)
#pragma unroll
      for (int tn = 0; tn < 2; ++tn)
        acc[tm][tn] = wmma_bf16(af[tm], bfr[tn], acc[tm][tn]);

    __syncthreads();  // reads done before next iteration's DMA overwrites
  }

  // Epilogue. C layout: lanes 0-15: row = vgpr i; lanes 16-31: row = 8+i; col = r.
#pragma unroll
  for (int tm = 0; tm < 2; ++tm)
#pragma unroll
    for (int tn = 0; tn < 2; ++tn)
#pragma unroll
      for (int i = 0; i < 8; ++i) {
        int row = blockRow + wm * 32 + tm * 16 + i + half * 8;
        int col = blockCol + wn * 32 + tn * 16 + r;
        float v = acc[tm][tn][i] + bias[col];
        if (EPI == 1) {
          v += res[(size_t)row * N + col];
          ((float*)Cout)[(size_t)row * N + col] = v;
        } else if (EPI == 3) {
          // scatter V into head-transposed layout Vt[b][h][e][s]
          int b = row >> 11, s = row & (S_ - 1);   // S = 2048
          int h = col >> 6,  e = col & 63;
          ((unsigned short*)Cout)[(((size_t)(b * H_ + h) * HS_ + e) << 11) + s] =
              f2bf(v);
        } else {
          if (EPI == 0) v *= scale;   // folds 1/sqrt(HS) into Q projection
          if (EPI == 2) v = 0.5f * v * (1.f + erff(v * 0.70710678118f));
          ((unsigned short*)Cout)[(size_t)row * N + col] = f2bf(v);
        }
      }
}

// ---------------------------------------------------------------------------
// One 32-key flash-attention block for a wave's 16 query rows.
// MASK=true only for the single diagonal block; bulk blocks skip all
// causal compares/selects. Scores arrive pre-scaled (Q holds q/sqrt(HS)).
// ---------------------------------------------------------------------------
template <bool MASK>
__device__ __forceinline__ void attn_block(
    int kb, int q0, int lane,
    const unsigned short* __restrict__ Kbh,  // K base for (b,h); + key*D_
    const unsigned short* __restrict__ Vh,   // Vt head slice: [64][S]
    const Frag (&qa)[2], v8f (&acc)[4],
    float (&mrow)[8], float (&lrow)[8],
    unsigned short (&P)[16][32]) {
  const int half = lane >> 4;
  const int r    = lane & 15;
  const float BIG = 3.0e38f;

  // ---- scores: S[16 x 32] = (Q/sqrt(HS)) K^T ----
  v8f sc[2];
#pragma unroll
  for (int t = 0; t < 2; ++t) {
    sc[t] = (v8f){0.f, 0.f, 0.f, 0.f, 0.f, 0.f, 0.f, 0.f};
    Frag kf0, kf1;
    const size_t krow = (size_t)(kb + t * 16 + r) * D_;
#pragma unroll
    for (int j = 0; j < 8; ++j) {
      kf0.u[j] = *(const unsigned int*)(Kbh + krow + kbaseB(j, half));
      kf1.u[j] = *(const unsigned int*)(Kbh + krow + 32 + kbaseB(j, half));
    }
    sc[t] = wmma_bf16(qa[0], kf0, sc[t]);
    sc[t] = wmma_bf16(qa[1], kf1, sc[t]);
  }

  // ---- online softmax ----
  float p0[8], p1[8], alph[8];
#pragma unroll
  for (int i = 0; i < 8; ++i) {
    float s0 = sc[0][i];
    float s1 = sc[1][i];
    if (MASK) {
      const int qrow = q0 + i + half * 8;
      if (kb + r > qrow) s0 = -BIG;
      if (kb + 16 + r > qrow) s1 = -BIG;
    }
    float v = fmaxf(s0, s1);
#pragma unroll
    for (int m = 1; m <= 8; m <<= 1) v = fmaxf(v, __shfl_xor(v, m, 32));
    const float mnew = fmaxf(mrow[i], v);
    const float e0 = __expf(s0 - mnew);
    const float e1 = __expf(s1 - mnew);
    float ps = e0 + e1;
#pragma unroll
    for (int m = 1; m <= 8; m <<= 1) ps += __shfl_xor(ps, m, 32);
    alph[i] = __expf(mrow[i] - mnew);
    lrow[i] = lrow[i] * alph[i] + ps;
    mrow[i] = mnew;
    p0[i] = e0; p1[i] = e1;
  }
#pragma unroll
  for (int t = 0; t < 4; ++t)
#pragma unroll
    for (int i = 0; i < 8; ++i) acc[t][i] *= alph[i];

  // ---- P: C layout -> A fragment via per-wave LDS (same-wave DS order) ----
#pragma unroll
  for (int i = 0; i < 8; ++i) {
    P[i + half * 8][r]      = f2bf(p0[i]);
    P[i + half * 8][16 + r] = f2bf(p1[i]);
  }
  Frag pf;
#pragma unroll
  for (int j = 0; j < 8; ++j)
    pf.u[j] = *(const unsigned int*)&P[r][kbaseA(j, half)];

  // ---- O += P @ V : V block 32x64 as 4 B-fragments, contiguous key pairs ----
#pragma unroll
  for (int t = 0; t < 4; ++t) {
    Frag vf;
#pragma unroll
    for (int j = 0; j < 8; ++j) {
      const int k = kbaseB(j, half);  // key within block (pair k, k+1)
      vf.u[j] = *(const unsigned int*)(Vh + (size_t)(t * 16 + r) * S_ + kb + k);
    }
    acc[t] = wmma_bf16(pf, vf, acc[t]);
  }
}

// ---------------------------------------------------------------------------
// Flash attention (causal). Q/K bf16 [B,S,H,HS] (Q pre-scaled by 1/sqrt(HS));
// V bf16 head-transposed [B,H,HS,S]. O bf16 [B,S,H,HS].
// Block = 128 thr = 4 waves; wave owns 16 query rows; key blocks of 32.
// ---------------------------------------------------------------------------
__global__ __launch_bounds__(128)
void k_attn(const unsigned short* __restrict__ Qb,
            const unsigned short* __restrict__ Kb,
            const unsigned short* __restrict__ Vt,
            unsigned short* __restrict__ Ob) {
  __shared__ __align__(16) unsigned short Plds[4][16][32];  // per-wave P scratch

  const int lane = threadIdx.x & 31;
  const int wave = threadIdx.x >> 5;
  const int half = lane >> 4;
  const int r    = lane & 15;
  const int bh   = blockIdx.y;       // b*H + h
  const int b    = bh >> 4;          // H = 16
  const int h    = bh & 15;
  const int q0   = (blockIdx.x * 4 + wave) * 16;

  const size_t hoff = (size_t)h * HS_;
  const unsigned short* Kbh = Kb + (size_t)b * S_ * D_ + hoff;
  const unsigned short* Vh  = Vt + ((size_t)(b * H_ + h) * HS_) * S_;  // [64][S]

  // Q fragments: two e-halves of [16 x 64]
  Frag qa[2];
#pragma unroll
  for (int eh = 0; eh < 2; ++eh)
#pragma unroll
    for (int j = 0; j < 8; ++j)
      qa[eh].u[j] = *(const unsigned int*)(Qb + ((size_t)(b * S_ + q0 + r)) * D_ +
                                           hoff + eh * 32 + kbaseA(j, half));

  float mrow[8], lrow[8];
  v8f acc[4];
#pragma unroll
  for (int t = 0; t < 4; ++t)
    acc[t] = (v8f){0.f, 0.f, 0.f, 0.f, 0.f, 0.f, 0.f, 0.f};
#pragma unroll
  for (int i = 0; i < 8; ++i) { mrow[i] = -3.0e38f; lrow[i] = 0.f; }

  // Blocks strictly below the diagonal need no causal mask; exactly one
  // (the 32-aligned block containing/preceding q0) needs masking.
  const int kdiag = (q0 >> 5) << 5;
  for (int kb = 0; kb < kdiag; kb += 32)
    attn_block<false>(kb, q0, lane, Kbh, Vh, qa, acc, mrow, lrow, Plds[wave]);
  attn_block<true>(kdiag, q0, lane, Kbh, Vh, qa, acc, mrow, lrow, Plds[wave]);

  // ---- normalize + store: O[b, q, h, :] bf16 ----
#pragma unroll
  for (int t = 0; t < 4; ++t)
#pragma unroll
    for (int i = 0; i < 8; ++i) {
      const int row = q0 + i + half * 8;
      const float o = acc[t][i] / lrow[i];
      Ob[((size_t)(b * S_ + row)) * D_ + hoff + t * 16 + r] = f2bf(o);
    }
}

// ---------------------------------------------------------------------------
// Host-side orchestration
// ---------------------------------------------------------------------------
extern "C" void kernel_launch(void* const* d_in, const int* in_sizes, int n_in,
                              void* d_out, int out_size, void* d_ws, size_t ws_size,
                              hipStream_t stream) {
  (void)in_sizes; (void)n_in; (void)out_size; (void)ws_size;
  const float* x     = (const float*)d_in[0];
  const float* Wq    = (const float*)d_in[1];
  const float* bq    = (const float*)d_in[2];
  const float* Wk    = (const float*)d_in[3];
  const float* bk    = (const float*)d_in[4];
  const float* Wv    = (const float*)d_in[5];
  const float* bv    = (const float*)d_in[6];
  const float* Wp    = (const float*)d_in[7];
  const float* bp    = (const float*)d_in[8];
  const float* W1    = (const float*)d_in[9];
  const float* b1    = (const float*)d_in[10];
  const float* W2    = (const float*)d_in[11];
  const float* b2    = (const float*)d_in[12];
  const float* gamma = (const float*)d_in[13];
  const float* beta  = (const float*)d_in[14];

  char* w = (char*)d_ws;
  const size_t MB = 1024ull * 1024ull;
  unsigned short* WqT = (unsigned short*)(w + 0 * MB);   // 2 MB  [1024][1024]
  unsigned short* WkT = (unsigned short*)(w + 2 * MB);   // 2 MB
  unsigned short* WvT = (unsigned short*)(w + 4 * MB);   // 2 MB
  unsigned short* WpT = (unsigned short*)(w + 6 * MB);   // 2 MB
  unsigned short* W1T = (unsigned short*)(w + 8 * MB);   // 8 MB  [4096][1024]
  unsigned short* W2T = (unsigned short*)(w + 16 * MB);  // 8 MB  [1024][4096]
  unsigned short* h_bf = (unsigned short*)(w + 24 * MB); // 8 MB (reused as h2)
  unsigned short* Q_bf = (unsigned short*)(w + 32 * MB); // 8 MB
  unsigned short* K_bf = (unsigned short*)(w + 40 * MB); // 8 MB
  unsigned short* Vt_bf = (unsigned short*)(w + 48 * MB);// 8 MB [B,H,HS,S]
  unsigned short* A_bf = (unsigned short*)(w + 56 * MB); // 8 MB
  float*          x1   = (float*)(w + 64 * MB);          // 16 MB
  unsigned short* G_bf = (unsigned short*)(w + 32 * MB); // 32 MB, reuses Q/K/V/A
  // total workspace footprint: 80 MB

  // weights -> transposed bf16 [N][K]
  k_cvt_head<<<4096, 256, 0, stream>>>(Wq, WqT);
  k_cvt_head<<<4096, 256, 0, stream>>>(Wk, WkT);
  k_cvt_head<<<4096, 256, 0, stream>>>(Wv, WvT);
  k_cvt_t<<<(D_ * D_) / 256, 256, 0, stream>>>(Wp, WpT, D_, 10);
  k_cvt_t<<<(D_ * DFF_) / 256, 256, 0, stream>>>(W1, W1T, D_, 12);
  k_cvt_t<<<(DFF_ * D_) / 256, 256, 0, stream>>>(W2, W2T, DFF_, 10);

  // h = LN(x)
  k_ln<<<BS_, 256, 0, stream>>>(x, gamma, beta, h_bf);

  // Q/K = h @ W + b  [BS, H*HS]; Q additionally scaled by 1/sqrt(HS);
  // V scattered to [B,H,HS,S]
  dim3 g1(D_ / 64, BS_ / 128);
  k_gemm<0><<<g1, 256, 0, stream>>>(h_bf, WqT, bq, nullptr, Q_bf, BS_, D_, D_,
                                    0.125f);
  k_gemm<0><<<g1, 256, 0, stream>>>(h_bf, WkT, bk, nullptr, K_bf, BS_, D_, D_,
                                    1.0f);
  k_gemm<3><<<g1, 256, 0, stream>>>(h_bf, WvT, bv, nullptr, Vt_bf, BS_, D_, D_,
                                    1.0f);

  // causal flash attention -> att [B,S,H,HS] == [BS, D]
  k_attn<<<dim3(S_ / 64, B_ * H_), 128, 0, stream>>>(Q_bf, K_bf, Vt_bf, A_bf);

  // x1 = x + att @ Wp + bp
  k_gemm<1><<<g1, 256, 0, stream>>>(A_bf, WpT, bp, x, x1, BS_, D_, D_, 1.0f);

  // h2 = LN(x1)
  k_ln<<<BS_, 256, 0, stream>>>(x1, gamma, beta, h_bf);

  // g = gelu(h2 @ W1 + b1)
  k_gemm<2><<<dim3(DFF_ / 64, BS_ / 128), 256, 0, stream>>>(
      h_bf, W1T, b1, nullptr, G_bf, BS_, DFF_, D_, 1.0f);

  // out = x1 + g @ W2 + b2
  k_gemm<1><<<g1, 256, 0, stream>>>(G_bf, W2T, b2, x1, (float*)d_out,
                                    BS_, D_, DFF_, 1.0f);
}
